// FlatHash7x7NNUEv1_47519518163396
// MI455X (gfx1250) — matrix operand order repaired and checked by reading, not compiled, not run on hardware
//
#include <hip/hip_runtime.h>

typedef __attribute__((ext_vector_type(16))) _Float16 v16h;
typedef __attribute__((ext_vector_type(8)))  _Float16 v8h;
typedef __attribute__((ext_vector_type(8)))  float    v8f;
typedef __attribute__((ext_vector_type(4)))  float    f32x4;

#define TABLE_MASK ((1 << 22) - 1)

// fake_quant(x, 128, 8) forward value
__device__ __forceinline__ float fq8(float x) {
    return fminf(fmaxf(rintf(x * 128.0f), -128.0f), 127.0f) * (1.0f / 128.0f);
}
// fake_quant(b, 128*128, 32) forward value (never clips for these magnitudes)
__device__ __forceinline__ float fq32b(float x) {
    return rintf(x * 16384.0f) * (1.0f / 16384.0f);
}

// Base-3 hash of a rotated 4x4 corner. digit == raw cell value (0/1/2).
// rot90 source mapping: k=0:(i,j) k=1:(j,3-i) k=2:(3-i,3-j) k=3:(3-j,i)
template <int K>
__device__ __forceinline__ int hash_corner(const int* __restrict__ cb, int y0, int x0) {
    constexpr int P3[16] = {1, 3, 9, 27, 81, 243, 729, 2187, 6561, 19683,
                            59049, 177147, 531441, 1594323, 4782969, 14348907};
    int s = 0;
#pragma unroll
    for (int p = 0; p < 16; ++p) {
        const int i = p >> 2, j = p & 3;
        int a, b;
        if (K == 0)      { a = i;     b = j;     }
        else if (K == 1) { a = j;     b = 3 - i; }
        else if (K == 2) { a = 3 - i; b = 3 - j; }
        else             { a = 3 - j; b = i;     }
        s += cb[(y0 + a) * 7 + (x0 + b)] * P3[p];
    }
    return s & TABLE_MASK;
}

// A-matrix (16x32 f16) fragment: lane L -> row M=L%16; halves 0..7 = K 8*(L/16)..+7,
// halves 8..15 = same +16.  (ISA 7.12.2, 16-bit A 16x32)
__device__ __forceinline__ v16h load_A(const _Float16* act, int rowBase, int lane) {
    const int M  = lane & 15;
    const int kb = (lane >> 4) << 3;   // 0 or 8
    const _Float16* p = act + (rowBase + M) * 32;
    v8h lo = *(const v8h*)(p + kb);
    v8h hi = *(const v8h*)(p + kb + 16);
    return __builtin_shufflevector(lo, hi, 0, 1, 2, 3, 4, 5, 6, 7, 8, 9, 10, 11, 12, 13, 14, 15);
}

// B-matrix (32x16 f16) fragment: lane L -> col N=L%16; halves 0..7 = K 8*(L/16)..+7,
// halves 8..15 = same +16 (matches load_A).
// B[k][n] = wq[n][k]  (v @ wq.T), so w's row-major layout is used directly.
__device__ __forceinline__ v16h load_B(const _Float16* w, int n, int lane) {
    const int kb = (lane >> 4) << 3;   // 0 or 8
    const _Float16* p = w + n * 32 + kb;
    v8h lo = *(const v8h*)(p);
    v8h hi = *(const v8h*)(p + 16);
    return __builtin_shufflevector(lo, hi, 0, 1, 2, 3, 4, 5, 6, 7, 8, 9, 10, 11, 12, 13, 14, 15);
}

__device__ __forceinline__ v8f wmma16(v16h a, v16h b, v8f c) {
    return __builtin_amdgcn_wmma_f32_16x16x32_f16(false, a, false, b, (short)0, c, false, false);
}

__global__ __launch_bounds__(256)
void nnue7x7_kernel(const int* __restrict__ cells, const float* __restrict__ table,
                    const float* __restrict__ w1, const float* __restrict__ b1,
                    const float* __restrict__ w2, const float* __restrict__ b2,
                    const float* __restrict__ w3, const float* __restrict__ b3,
                    float* __restrict__ out, int B) {
    __shared__ __align__(16) _Float16 act[128 * 32];   // per-row activations (f16, exact 1/128 grid at L1 input)
    __shared__ __align__(16) _Float16 wq1[32 * 32];
    __shared__ __align__(16) _Float16 wq2[32 * 32];
    __shared__ __align__(16) _Float16 wq3[16 * 32];    // rows n>=3 zero
    __shared__ float bq1[32], bq2[32], bq3[16];
    __shared__ int   idxs[128][4];

    const int t     = threadIdx.x;
    const int eBase = blockIdx.x * 128;

    // ---- Phase 1a: quantize weights/biases into LDS ----
    for (int i = t; i < 1024; i += 256) {
        wq1[i] = (_Float16)fq8(w1[i]);
        wq2[i] = (_Float16)fq8(w2[i]);
    }
    for (int i = t; i < 512; i += 256)
        wq3[i] = (i < 96) ? (_Float16)fq8(w3[i]) : (_Float16)0.0f;
    if (t < 32) { bq1[t] = fq32b(b1[t]); bq2[t] = fq32b(b2[t]); }
    if (t < 16) bq3[t] = (t < 3) ? fq32b(b3[t]) : 0.0f;

    // ---- Phase 1b: hash indices (even threads, one element each) + row prefetch ----
    // Table is 512 MB (> 192 MB L2) with random, reuse-free access: prefetch the 4
    // rows now so the HBM latency overlaps the barrier + feature-store phase.
    if ((t & 1) == 0) {
        const int e = t >> 1;
        const int* cb = cells + (size_t)(eBase + e) * 49;
        idxs[e][0] = hash_corner<0>(cb, 0, 0);
        idxs[e][1] = hash_corner<1>(cb, 0, 3);
        idxs[e][2] = hash_corner<2>(cb, 3, 3);
        idxs[e][3] = hash_corner<3>(cb, 3, 0);
#pragma unroll
        for (int c = 0; c < 4; ++c) {
            const char* p = (const char*)(table + (size_t)idxs[e][c] * 32);
            __builtin_prefetch(p, 0, 3);        // 128B row = 2 x 64B
            __builtin_prefetch(p + 64, 0, 3);
        }
    }
    __syncthreads();

    // ---- Phase 2: gather + int8 fake-quant + sum -> clipped f16 feat in LDS ----
    {
        const int e  = t >> 1;
        const int f0 = (t & 1) * 16;   // lane pairs cover one 128B table row coalesced
        float acc[16];
#pragma unroll
        for (int i = 0; i < 16; ++i) acc[i] = 0.0f;
#pragma unroll
        for (int c = 0; c < 4; ++c) {
            const f32x4* rp = (const f32x4*)(table + (size_t)idxs[e][c] * 32 + f0);
#pragma unroll
            for (int q = 0; q < 4; ++q) {
                f32x4 v = __builtin_nontemporal_load(rp + q);  // stream: don't pollute L2
#pragma unroll
                for (int i = 0; i < 4; ++i) acc[q * 4 + i] += fq8(v[i]);
            }
        }
        v8h h0, h1;
#pragma unroll
        for (int i = 0; i < 8; ++i) {
            h0[i] = (_Float16)(fminf(fmaxf(acc[i],     -1.0f), 127.0f / 128.0f));
            h1[i] = (_Float16)(fminf(fmaxf(acc[i + 8], -1.0f), 127.0f / 128.0f));
        }
        *(v8h*)(act + e * 32 + f0)     = h0;
        *(v8h*)(act + e * 32 + f0 + 8) = h1;
    }
    __syncthreads();

    // ---- Phase 3: WMMA MLP, each wave owns 16 batch rows ----
    const int lane    = t & 31;
    const int rowBase = (t >> 5) * 16;
    const int n       = lane & 15;
    const int mHalf   = (lane >> 4) << 3;

    // Layer 1: 32->32, output clip [0, 127/128]
    {
        v16h a  = load_A(act, rowBase, lane);
        v16h B0 = load_B(wq1, n, lane);
        v16h B1 = load_B(wq1, n + 16, lane);
        v8f c0 = {}, c1 = {};
        c0 = wmma16(a, B0, c0);
        c1 = wmma16(a, B1, c1);
#pragma unroll
        for (int r = 0; r < 8; ++r) {
            const int M = rowBase + r + mHalf;
            float o0 = fminf(fmaxf(c0[r] + bq1[n],      0.0f), 127.0f / 128.0f);
            float o1 = fminf(fmaxf(c1[r] + bq1[n + 16], 0.0f), 127.0f / 128.0f);
            act[M * 32 + n]      = (_Float16)o0;
            act[M * 32 + n + 16] = (_Float16)o1;
        }
    }
    __syncthreads();

    // Layer 2: 32->32, output clip [0, 127/128]
    {
        v16h a  = load_A(act, rowBase, lane);
        v16h B0 = load_B(wq2, n, lane);
        v16h B1 = load_B(wq2, n + 16, lane);
        v8f c0 = {}, c1 = {};
        c0 = wmma16(a, B0, c0);
        c1 = wmma16(a, B1, c1);
#pragma unroll
        for (int r = 0; r < 8; ++r) {
            const int M = rowBase + r + mHalf;
            float o0 = fminf(fmaxf(c0[r] + bq2[n],      0.0f), 127.0f / 128.0f);
            float o1 = fminf(fmaxf(c1[r] + bq2[n + 16], 0.0f), 127.0f / 128.0f);
            act[M * 32 + n]      = (_Float16)o0;
            act[M * 32 + n + 16] = (_Float16)o1;
        }
    }
    __syncthreads();

    // Layer 3: 32->3 (single N-tile, cols 3..15 zero-weighted)
    {
        v16h a  = load_A(act, rowBase, lane);
        v16h B0 = load_B(wq3, n, lane);
        v8f c0 = {};
        c0 = wmma16(a, B0, c0);
        if (n < 3) {
#pragma unroll
            for (int r = 0; r < 8; ++r) {
                const int row = eBase + rowBase + r + mHalf;
                __builtin_nontemporal_store(c0[r] + bq3[n], out + (size_t)row * 3 + n);
            }
        }
    }

    // ---- Phase 4: zero this block's policy slice (write-once stream) ----
    {
        float* pol = out + (size_t)B * 3 + (size_t)eBase * 49;
        for (int i = t; i < 128 * 49; i += 256)
            __builtin_nontemporal_store(0.0f, pol + i);
    }
}

extern "C" void kernel_launch(void* const* d_in, const int* in_sizes, int n_in,
                              void* d_out, int out_size, void* d_ws, size_t ws_size,
                              hipStream_t stream) {
    const int*   cells = (const int*)d_in[0];
    const float* table = (const float*)d_in[1];
    const float* w1    = (const float*)d_in[2];
    const float* b1    = (const float*)d_in[3];
    const float* w2    = (const float*)d_in[4];
    const float* b2    = (const float*)d_in[5];
    const float* w3    = (const float*)d_in[6];
    const float* b3    = (const float*)d_in[7];
    float* out = (float*)d_out;

    const int B = in_sizes[0] / 49;          // 65536
    const int blocks = B / 128;              // 128 batch rows per block (8 waves x 16)
    nnue7x7_kernel<<<blocks, 256, 0, stream>>>(cells, table, w1, b1, w2, b2, w3, b3, out, B);
}
